// HTFE_73272142070186
// MI455X (gfx1250) — matrix operand order repaired and checked by reading, not compiled, not run on hardware
//
#include <hip/hip_runtime.h>
#include <hip/hip_fp16.h>

typedef __attribute__((ext_vector_type(16))) _Float16 v16h;
typedef __attribute__((ext_vector_type(8)))  _Float16 v8h;
typedef __attribute__((ext_vector_type(8)))  float    v8f;

#define B_  64
#define L_  4096
#define C_  16
#define PL_ 128
#define PC_ 8
#define NP_ 64
#define D_  64
#define T_  0.01f          // 1/FS, FS=100

// Fast silu: x * rcp(1+e^-x) -> v_exp_f32 + v_rcp_f32, no precise-div chain.
__device__ __forceinline__ float silu_f(float v) {
    return v * __builtin_amdgcn_rcpf(1.0f + __expf(-v));
}

// ---------------------------------------------------------------------------
// Kernel 0: W3 (8192 x 64 f32, row-major) -> W3T (64 x 8192 f16) so stage-B
// B-fragments are contiguous along K (two 16B loads per lane).
// ---------------------------------------------------------------------------
__global__ __launch_bounds__(256) void htfe_w3_convert(
    const float* __restrict__ W3, _Float16* __restrict__ W3T)
{
    int tid = blockIdx.x * 256 + threadIdx.x;   // 0 .. 8192*64-1
    int n = tid & 63;
    int k = tid >> 6;
    W3T[(size_t)n * (PL_ * D_) + k] = (_Float16)W3[(size_t)k * D_ + n];
}

// ---------------------------------------------------------------------------
// Stage A: per patch, gather feats (128 x 17, K zero-padded to 32) and compute
// h = silu(feats @ W1 + b1) -> (128 x 64) f16 into hbuf.
// One block per patch, 8 waves; wave w = M-tile, 4 N-tiles (4 WMMAs).
// A-fragment (v_wmma_f32_16x16x32_f16, wave32):
//   lanes 0-15 : row M=r, K={0..7} (x gathers)      and {16..23} (c[7], pad)
//   lanes 16-31: row M=r, K={8..15} (t, c[0..6])    and {24..31} (pad)
// ---------------------------------------------------------------------------
__global__ __launch_bounds__(256) void htfe_stageA(
    const float* __restrict__ x,   const int* __restrict__ startL,
    const int* __restrict__ idxC,  const float* __restrict__ W1,
    const float* __restrict__ b1,  _Float16* __restrict__ hbuf)
{
    const int patch = blockIdx.x;              // 0 .. B*NP-1
    const int b     = patch >> 6;              // / NP (NP = 64)
    const int wave  = threadIdx.x >> 5;
    const int lane  = threadIdx.x & 31;
    const int r     = lane & 15;
    const int hs    = lane >> 4;
    const int m0    = wave * 16;

    const int sL = startL[patch];
    int ic[PC_];
    #pragma unroll
    for (int c = 0; c < PC_; ++c) ic[c] = idxC[patch * PC_ + c];

    const int lL = (sL + m0 + r) & (L_ - 1);
    const float* xrow = x + ((size_t)b * L_ + lL) * C_;

    v16h a = {};
    if (hs == 0) {
        #pragma unroll
        for (int k = 0; k < 8; ++k) a[k] = (_Float16)xrow[ic[k]]; // K=0..7
        a[8] = (_Float16)(float)ic[7];                            // K=16 (c[7])
    } else {
        a[0] = (_Float16)((float)lL * T_);                        // K=8  (t)
        #pragma unroll
        for (int c = 0; c < 7; ++c) a[1 + c] = (_Float16)(float)ic[c]; // K=9..15
    }

    #pragma unroll
    for (int j = 0; j < 4; ++j) {
        const int n0  = j * 16;
        const int col = n0 + r;
        v16h bf = {};
        if (hs == 0) {
            #pragma unroll
            for (int k = 0; k < 8; ++k) bf[k] = (_Float16)W1[k * D_ + col];
            bf[8] = (_Float16)W1[16 * D_ + col];   // only K=16 is a valid row
        } else {
            #pragma unroll
            for (int k = 0; k < 8; ++k) bf[k] = (_Float16)W1[(8 + k) * D_ + col];
        }
        v8f acc = {};
        acc = __builtin_amdgcn_wmma_f32_16x16x32_f16(
            false, a, false, bf, (short)0, acc, false, false);

        const float bias = b1[col];
        #pragma unroll
        for (int v = 0; v < 8; ++v) {
            const int row = m0 + v + 8 * hs;       // C/D layout: VGPR v -> M
            hbuf[((size_t)patch * PL_ + row) * D_ + col] =
                (_Float16)silu_f(acc[v] + bias);
        }
    }
}

// ---------------------------------------------------------------------------
// Stage B: out(4096 x 64) = silu( h(4096 x 8192) @ W3(8192 x 64) + b3 )
// 128 blocks x (M=32 rows); 4 waves = 4 N-tiles; each wave does 2 WMMAs per
// K-step (two M-tiles sharing one B-fragment).
// Shared 32x32-half A-tile staged via async global->LDS (ASYNCcnt path),
// double-buffered; B-fragments stream straight from global (per-wave unique).
// ---------------------------------------------------------------------------
__global__ __launch_bounds__(128) void htfe_stageB(
    const _Float16* __restrict__ hbuf, const _Float16* __restrict__ W3T,
    const float* __restrict__ b3,      float* __restrict__ out)
{
    __shared__ __align__(16) _Float16 smA[2][32][32];   // 2 x 2KB double buffer

    const int wave = threadIdx.x >> 5;
    const int lane = threadIdx.x & 31;
    const int r    = lane & 15;
    const int hs   = lane >> 4;
    const int m0   = blockIdx.x * 32;
    const int n0   = wave * 16;
    const int K    = PL_ * D_;                          // 8192
    const int NIT  = K / 32;                            // 256 K-steps

    // Cooperative async copy: 128 threads x 16B = one 32x32-half tile.
    const int chunk = threadIdx.x;                      // 0..127
    const int crow  = chunk >> 2;                       // 0..31
    const int ccol  = (chunk & 3) * 8;                  // halves
    const _Float16* gsrc = hbuf + (size_t)(m0 + crow) * K + ccol;
    const unsigned lds0 = (unsigned)(uintptr_t)&smA[0][crow][ccol];
    const unsigned lds1 = (unsigned)(uintptr_t)&smA[1][crow][ccol];

    const _Float16* brow = W3T + (size_t)(n0 + r) * K + (hs ? 8 : 0);

    v8f acc0 = {}, acc1 = {};

    // Prologue: stage K-step 0 into buffer 0 (one vector async op per wave).
    asm volatile("global_load_async_to_lds_b128 %0, %1, off"
                 :: "v"(lds0), "v"(gsrc) : "memory");

    for (int i = 0; i < NIT; ++i) {
        const int k0 = i * 32;
        // All of this wave's async issues complete (in-order) ...
        asm volatile("s_wait_asynccnt 0" ::: "memory");
        __syncthreads();   // ... and every wave's quarter of the tile landed.

        // Issue next tile into the other buffer before computing (hides copy
        // behind this step's WMMAs; buffer freed by the barrier above).
        if (i + 1 < NIT) {
            const unsigned dst = ((i + 1) & 1) ? lds1 : lds0;
            asm volatile("global_load_async_to_lds_b128 %0, %1, off"
                         :: "v"(dst), "v"(gsrc + (size_t)(i + 1) * 32)
                         : "memory");
        }

        // B fragment: K={8h..8h+7} and {16+8h..16+8h+7} from transposed W3.
        v8h blo = *(const v8h*)(brow + k0);
        v8h bhi = *(const v8h*)(brow + k0 + 16);
        v16h bf;
        #pragma unroll
        for (int e = 0; e < 8; ++e) { bf[e] = blo[e]; bf[8 + e] = bhi[e]; }

        // Two A fragments (M-tiles) from the staged LDS tile (ds_load_b128).
        const _Float16* sa = &smA[i & 1][0][0];
        const int kb = hs * 8;
        v8h a0lo = *(const v8h*)(sa + (size_t)(r)      * 32 + kb);
        v8h a0hi = *(const v8h*)(sa + (size_t)(r)      * 32 + kb + 16);
        v8h a1lo = *(const v8h*)(sa + (size_t)(16 + r) * 32 + kb);
        v8h a1hi = *(const v8h*)(sa + (size_t)(16 + r) * 32 + kb + 16);
        v16h a0, a1;
        #pragma unroll
        for (int e = 0; e < 8; ++e) {
            a0[e] = a0lo[e]; a0[8 + e] = a0hi[e];
            a1[e] = a1lo[e]; a1[8 + e] = a1hi[e];
        }

        acc0 = __builtin_amdgcn_wmma_f32_16x16x32_f16(
            false, a0, false, bf, (short)0, acc0, false, false);
        acc1 = __builtin_amdgcn_wmma_f32_16x16x32_f16(
            false, a1, false, bf, (short)0, acc1, false, false);
    }

    const int col  = n0 + r;
    const float bias = b3[col];
    #pragma unroll
    for (int v = 0; v < 8; ++v) {
        const int row = m0 + v + 8 * hs;
        out[(size_t)row       * D_ + col] = silu_f(acc0[v] + bias);
        out[(size_t)(row + 16) * D_ + col] = silu_f(acc1[v] + bias);
    }
}

// ---------------------------------------------------------------------------
// Inputs (setup_inputs order): x, start_L, idx_C, W1, b1, W2, b2, W3, b3.
// W2/b2 feed dead code (unused FFT branch) in the reference.
// Workspace: hbuf f16 (64 MB) + W3T f16 (1 MB).
// ---------------------------------------------------------------------------
extern "C" void kernel_launch(void* const* d_in, const int* in_sizes, int n_in,
                              void* d_out, int out_size, void* d_ws, size_t ws_size,
                              hipStream_t stream) {
    const float* x      = (const float*)d_in[0];
    const int*   startL = (const int*)  d_in[1];
    const int*   idxC   = (const int*)  d_in[2];
    const float* W1     = (const float*)d_in[3];
    const float* b1     = (const float*)d_in[4];
    const float* W3     = (const float*)d_in[7];
    const float* b3     = (const float*)d_in[8];
    float* out = (float*)d_out;

    _Float16* hbuf = (_Float16*)d_ws;                              // 33,554,432 halves
    _Float16* W3T  = hbuf + (size_t)(B_ * NP_) * PL_ * D_;         // 524,288 halves

    htfe_w3_convert<<<(PL_ * D_ * D_) / 256, 256, 0, stream>>>(W3, W3T);
    htfe_stageA<<<B_ * NP_, 256, 0, stream>>>(x, startL, idxC, W1, b1, hbuf);
    htfe_stageB<<<(B_ * NP_) / 32, 128, 0, stream>>>(hbuf, W3T, b3, out);
}
